// C_LMCL_36782099923414
// MI455X (gfx1250) — matrix operand order repaired and checked by reading, not compiled, not run on hardware
//
#include <hip/hip_runtime.h>

typedef __attribute__((ext_vector_type(16))) __bf16 v16bf;
typedef __attribute__((ext_vector_type(8)))  __bf16 v8bf;
typedef __attribute__((ext_vector_type(8)))  float  v8f;

constexpr int   Bsz  = 512;
constexpr int   Edim = 512;
constexpr int   Ccls = 100000;
constexpr int   CT   = 48;                       // classes per block = 3 WMMA N-tiles
constexpr int   NBLK = (Ccls + CT - 1) / CT;     // 2084
constexpr int   LDW  = Edim + 16;                // padded bf16 row (keeps 32B align, shifts banks)
constexpr float Sc = 30.0f, Mg = 0.35f, LAM = 0.01f, ALPH = 0.5f, EPSF = 1e-6f;

// ---- workspace layout (byte offsets) ----
constexpr size_t OFF_EMBN = 0;                                  // Bsz*Edim bf16
constexpr size_t OFF_PMAX = OFF_EMBN + (size_t)Bsz * Edim * 2;  // Bsz*NBLK f32
constexpr size_t OFF_PSUM = OFF_PMAX + (size_t)Bsz * NBLK * 4;
constexpr size_t OFF_PAV  = OFF_PSUM + (size_t)Bsz * NBLK * 4;
constexpr size_t OFF_PAI  = OFF_PAV  + (size_t)Bsz * NBLK * 4;  // int
constexpr size_t OFF_LLG  = OFF_PAI  + (size_t)Bsz * NBLK * 4;  // Bsz f32 (m_logit at label)
constexpr size_t OFF_APP  = OFF_LLG  + (size_t)Bsz * 4;         // Bsz f32 (appear count)
constexpr size_t OFF_ACC  = OFF_APP  + (size_t)Bsz * 4;         // 3 f32: loss, prec, closs

__global__ __launch_bounds__(32) void k_init(float* acc) {
  if (threadIdx.x < 3) acc[threadIdx.x] = 0.0f;
}

// Row-normalize embeddings -> bf16 (one block per row)
__global__ __launch_bounds__(256) void k_embnorm(const float* __restrict__ emb,
                                                 __bf16* __restrict__ embn) {
  int b = blockIdx.x, t = threadIdx.x;
  __shared__ float red[256];
  float2 v = ((const float2*)(emb + (size_t)b * Edim))[t];
  red[t] = v.x * v.x + v.y * v.y;
  __syncthreads();
  for (int s = 128; s > 0; s >>= 1) {
    if (t < s) red[t] += red[t + s];
    __syncthreads();
  }
  float rn = rsqrtf(red[0]);
  union { __bf16 h[2]; unsigned u; } pk;
  pk.h[0] = (__bf16)(v.x * rn);
  pk.h[1] = (__bf16)(v.y * rn);
  ((unsigned*)embn)[(size_t)b * (Edim / 2) + t] = pk.u;
}

// Fused: stage weight tile (bf16, LDS) + rsq norms, WMMA GEMM, margin,
// online-softmax/argmax scan -> per-(row,block) partials.
__global__ __launch_bounds__(256) void k_gemm(const __bf16* __restrict__ embn,
                                              const float*  __restrict__ weights,
                                              const int*    __restrict__ label,
                                              float* __restrict__ pmax,
                                              float* __restrict__ psum,
                                              float* __restrict__ paval,
                                              int*   __restrict__ paidx,
                                              float* __restrict__ labellogit) {
  __shared__ __bf16 wtile[CT * LDW];          // 50,688 B
  __shared__ float  snorm[CT];                // S / ||w_c||
  __shared__ float  scanbuf[8][16][16];       // 8 KB, per-wave 16x16 f32 tile spill

  const int t    = threadIdx.x;
  const int lane = t & 31;
  const int wave = t >> 5;
  const int c0   = blockIdx.x * CT;

  // ---- stage weights: each wave owns rows wave, wave+8, ... (6 rows) ----
  for (int r = wave; r < CT; r += 8) {
    int c = c0 + r;
    float ss = 0.0f;
    union { v16bf v; __bf16 h[16]; } pk;
    if (c < Ccls) {
      const float4* wrow = (const float4*)(weights + (size_t)c * Edim);
      #pragma unroll
      for (int q = 0; q < 4; ++q) {
        float4 w = wrow[lane * 4 + q];
        ss += w.x * w.x + w.y * w.y + w.z * w.z + w.w * w.w;
        pk.h[q * 4 + 0] = (__bf16)w.x;
        pk.h[q * 4 + 1] = (__bf16)w.y;
        pk.h[q * 4 + 2] = (__bf16)w.z;
        pk.h[q * 4 + 3] = (__bf16)w.w;
      }
    } else {
      #pragma unroll
      for (int j = 0; j < 16; ++j) pk.h[j] = (__bf16)0.0f;
    }
    *(v16bf*)&wtile[r * LDW + lane * 16] = pk.v;
    #pragma unroll
    for (int off = 16; off > 0; off >>= 1) ss += __shfl_xor(ss, off, 32);
    if (lane == 0) snorm[r] = ss;
  }
  __syncthreads();
  if (t < CT) {
    float s = snorm[t];
    snorm[t] = (s > 0.0f) ? Sc * rsqrtf(s) : 0.0f;   // fold CosFace scale into column norm
  }
  __syncthreads();

  // ---- WMMA phase: wave handles M-tiles {wave, wave+8, wave+16, wave+24} ----
  const int m    = lane & 15;
  const int hi   = lane >> 4;
  const int kA   = hi * 16;   // A k order used here (contiguous): lanes 16-31 at K+16..31
  const int kB   = hi * 16;   // B k order used here (contiguous): lanes 16-31 at K+16..31
  const int col0 = hi * 8;    // epilogue: this lane-half scans cols col0..col0+7

  for (int mt = wave; mt < 32; mt += 8) {
    const int bg = mt * 16 + m;                 // global batch row for this lane
    const __bf16* arow = embn + (size_t)bg * Edim;
    v8f acc0 = {}, acc1 = {}, acc2 = {};

    for (int ks = 0; ks < 16; ++ks) {
      const int k0 = ks * 32;
      union { v16bf v; v8bf h[2]; } af;
      af.h[0] = *(const v8bf*)(arow + k0 + kA);
      af.h[1] = *(const v8bf*)(arow + k0 + kA + 8);
      v16bf b0 = *(const v16bf*)&wtile[(0 * 16 + m) * LDW + k0 + kB];
      v16bf b1 = *(const v16bf*)&wtile[(1 * 16 + m) * LDW + k0 + kB];
      v16bf b2 = *(const v16bf*)&wtile[(2 * 16 + m) * LDW + k0 + kB];
      acc0 = __builtin_amdgcn_wmma_f32_16x16x32_bf16(false, af.v, false, b0,
                                                     (short)0, acc0, false, false);
      acc1 = __builtin_amdgcn_wmma_f32_16x16x32_bf16(false, af.v, false, b1,
                                                     (short)0, acc1, false, false);
      acc2 = __builtin_amdgcn_wmma_f32_16x16x32_bf16(false, af.v, false, b2,
                                                     (short)0, acc2, false, false);
    }

    // ---- epilogue: margin + online softmax + argmax over this block's 48 cols.
    // Each lane-half scans 8 of the 16 columns of its row; halves merged via shfl.
    float rmax = -3.0e38f, rsum = 0.0f, aval = -3.0e38f;
    int   aidx = 0x7fffffff;
    const int   lab = label[bg];
    const int   rb  = hi * 8;
    const float mS  = Sc * Mg;

    #pragma unroll
    for (int nt = 0; nt < 3; ++nt) {
      const v8f a = (nt == 0) ? acc0 : ((nt == 1) ? acc1 : acc2);
      const float rn = snorm[nt * 16 + m];      // this lane's column: S/||w||
      #pragma unroll
      for (int r = 0; r < 8; ++r)
        scanbuf[wave][rb + r][m] = a[r] * rn;   // wave-private; DS in-order
      for (int cc = 0; cc < 8; ++cc) {
        int col = col0 + cc;
        int c   = c0 + nt * 16 + col;
        if (c < Ccls) {
          float ml = scanbuf[wave][m][col] - ((c == lab) ? mS : 0.0f);
          if (c == lab) labellogit[bg] = ml;    // exactly one lane owns this column
          if (ml > aval) { aval = ml; aidx = c; }
          if (ml > rmax) {
            rsum = rsum * __expf(rmax - ml) + 1.0f;
            rmax = ml;
          } else {
            rsum += __expf(ml - rmax);
          }
        }
      }
    }

    // merge the two lane-halves (partner = lane ^ 16)
    float omax = __shfl_xor(rmax, 16, 32);
    float osum = __shfl_xor(rsum, 16, 32);
    float oav  = __shfl_xor(aval, 16, 32);
    int   oai  = __shfl_xor(aidx, 16, 32);
    float nmax = fmaxf(rmax, omax);
    rsum = rsum * __expf(rmax - nmax) + osum * __expf(omax - nmax);
    rmax = nmax;
    if (oav > aval || (oav == aval && oai < aidx)) { aval = oav; aidx = oai; }

    if (lane < 16) {
      size_t pi = (size_t)bg * NBLK + blockIdx.x;
      pmax[pi]  = rmax;
      psum[pi]  = rsum;
      paval[pi] = aval;
      paidx[pi] = aidx;
    }
  }
}

// One block per batch row: merge 2084 partials -> lse, label logp, top-1
__global__ __launch_bounds__(256) void k_rowreduce(const float* __restrict__ pmax,
                                                   const float* __restrict__ psum,
                                                   const float* __restrict__ paval,
                                                   const int*   __restrict__ paidx,
                                                   const float* __restrict__ labellogit,
                                                   const int*   __restrict__ label,
                                                   float* __restrict__ accums) {
  int b = blockIdx.x, t = threadIdx.x;
  __shared__ float sm[256];
  __shared__ float ssum[256];
  __shared__ float sav[256];
  __shared__ int   sai[256];

  float mx = -3.0e38f;
  for (int j = t; j < NBLK; j += 256) mx = fmaxf(mx, pmax[(size_t)b * NBLK + j]);
  sm[t] = mx;
  __syncthreads();
  for (int s = 128; s > 0; s >>= 1) {
    if (t < s) sm[t] = fmaxf(sm[t], sm[t + s]);
    __syncthreads();
  }
  float gmax = sm[0];
  __syncthreads();

  float sum = 0.0f, av = -3.0e38f;
  int ai = 0x7fffffff;
  for (int j = t; j < NBLK; j += 256) {
    size_t i = (size_t)b * NBLK + j;
    sum += psum[i] * __expf(pmax[i] - gmax);
    float v = paval[i];
    int   x = paidx[i];
    if (v > av || (v == av && x < ai)) { av = v; ai = x; }
  }
  ssum[t] = sum; sav[t] = av; sai[t] = ai;
  __syncthreads();
  for (int s = 128; s > 0; s >>= 1) {
    if (t < s) {
      ssum[t] += ssum[t + s];
      if (sav[t + s] > sav[t] || (sav[t + s] == sav[t] && sai[t + s] < sai[t])) {
        sav[t] = sav[t + s];
        sai[t] = sai[t + s];
      }
    }
    __syncthreads();
  }
  if (t == 0) {
    float lse = gmax + __logf(ssum[0]);
    float lp  = labellogit[b] - lse;
    unsafeAtomicAdd(&accums[0], -lp);
    if (sai[0] == label[b]) unsafeAtomicAdd(&accums[1], 1.0f);
  }
}

// new_centers = centers  (d_out+2 is only 8B-aligned -> float2 copies)
__global__ __launch_bounds__(256) void k_copycenters(const float* __restrict__ cen,
                                                     float* __restrict__ outc) {
  const size_t n = (size_t)Ccls * Edim / 2;
  const float2* s = (const float2*)cen;
  float2*       d = (float2*)outc;
  for (size_t i = (size_t)blockIdx.x * blockDim.x + threadIdx.x; i < n;
       i += (size_t)gridDim.x * blockDim.x)
    d[i] = s[i];
}

__global__ __launch_bounds__(512) void k_appear(const int* __restrict__ label,
                                                float* __restrict__ appear) {
  __shared__ int sl[Bsz];
  int t = threadIdx.x;
  sl[t] = label[t];
  __syncthreads();
  int my = sl[t], cnt = 0;
  for (int j = 0; j < Bsz; ++j) cnt += (sl[j] == my);
  appear[t] = (float)cnt;
}

// Center MSE + scatter update: outc[lab] += ALPHA*(emb - centers[lab])/(cnt+eps)
__global__ __launch_bounds__(256) void k_centerupd(const float* __restrict__ emb,
                                                   const float* __restrict__ cen,
                                                   const int*   __restrict__ label,
                                                   const float* __restrict__ appear,
                                                   float* __restrict__ outc,
                                                   float* __restrict__ accums) {
  int b = blockIdx.x, t = threadIdx.x;
  int lab = label[b];
  __shared__ float red[256];
  float2 e = ((const float2*)(emb + (size_t)b * Edim))[t];
  float2 c = ((const float2*)(cen + (size_t)lab * Edim))[t];
  float dx = e.x - c.x, dy = e.y - c.y;
  red[t] = dx * dx + dy * dy;
  __syncthreads();
  for (int s = 128; s > 0; s >>= 1) {
    if (t < s) red[t] += red[t + s];
    __syncthreads();
  }
  if (t == 0) unsafeAtomicAdd(&accums[2], red[0]);
  float scale = ALPH / (appear[b] + EPSF);
  unsafeAtomicAdd(&outc[(size_t)lab * Edim + 2 * t + 0], dx * scale);
  unsafeAtomicAdd(&outc[(size_t)lab * Edim + 2 * t + 1], dy * scale);
}

__global__ __launch_bounds__(32) void k_finalize(const float* __restrict__ accums,
                                                 float* __restrict__ out) {
  if (threadIdx.x == 0) {
    out[0] = accums[1] * (100.0f / (float)Bsz);
    out[1] = accums[0] / (float)Bsz +
             LAM * accums[2] / ((float)Bsz * (float)Edim);
  }
}

extern "C" void kernel_launch(void* const* d_in, const int* in_sizes, int n_in,
                              void* d_out, int out_size, void* d_ws, size_t ws_size,
                              hipStream_t stream) {
  (void)in_sizes; (void)n_in; (void)out_size; (void)ws_size;
  const float* emb = (const float*)d_in[0];
  const float* wts = (const float*)d_in[1];
  const float* cen = (const float*)d_in[2];
  const int*   lab = (const int*)d_in[3];

  char* ws = (char*)d_ws;
  __bf16* embn   = (__bf16*)(ws + OFF_EMBN);
  float*  pmax   = (float*)(ws + OFF_PMAX);
  float*  psum   = (float*)(ws + OFF_PSUM);
  float*  paval  = (float*)(ws + OFF_PAV);
  int*    paidx  = (int*)  (ws + OFF_PAI);
  float*  llogit = (float*)(ws + OFF_LLG);
  float*  appear = (float*)(ws + OFF_APP);
  float*  accums = (float*)(ws + OFF_ACC);

  float* out  = (float*)d_out;
  float* outc = out + 2;

  k_init<<<1, 32, 0, stream>>>(accums);
  k_embnorm<<<Bsz, 256, 0, stream>>>(emb, embn);
  k_gemm<<<NBLK, 256, 0, stream>>>(embn, wts, lab, pmax, psum, paval, paidx, llogit);
  k_rowreduce<<<Bsz, 256, 0, stream>>>(pmax, psum, paval, paidx, llogit, lab, accums);
  k_copycenters<<<4096, 256, 0, stream>>>(cen, outc);
  k_appear<<<1, Bsz, 0, stream>>>(lab, appear);
  k_centerupd<<<Bsz, 256, 0, stream>>>(emb, cen, lab, appear, outc, accums);
  k_finalize<<<1, 32, 0, stream>>>(accums, out);
}
